// VectorQuantizerEMA_26972394619048
// MI455X (gfx1250) — matrix operand order repaired and checked by reading, not compiled, not run on hardware
//
#include <hip/hip_runtime.h>
#include <hip/hip_bf16.h>
#include <math.h>

#define N_ROWS 65536
#define K_EMB  2048
#define D_DIM  64
#define DECAYF 0.99f
#define OMDF   0.01f
#define EPSV   1e-5f

typedef __attribute__((ext_vector_type(16))) __bf16 v16bf;
typedef __attribute__((ext_vector_type(8)))  float  v8f;
typedef __attribute__((ext_vector_type(4)))  float  v4f;

// LDS layout: bf16 embedding rows, 64 elems = 32 dwords, padded to 36 dwords
// (144B) so lanes 0..15 reading 16 consecutive rows hit 16 distinct banks.
#define ROW_U   36
#define ESQ_OFF (K_EMB * ROW_U)  // dword index where f32 e_sq[] region starts

__device__ __forceinline__ unsigned int pack_bf16x2(float lo, float hi) {
    __bf16 a = (__bf16)lo, b = (__bf16)hi;
    unsigned short ua = __builtin_bit_cast(unsigned short, a);
    unsigned short ub = __builtin_bit_cast(unsigned short, b);
    return (unsigned int)ua | ((unsigned int)ub << 16);
}

union Frag { v16bf v; unsigned int u[8]; };

__global__ void vq_init(float* ws, int n) {
    int i = blockIdx.x * blockDim.x + threadIdx.x;
    if (i < n) ws[i] = 0.0f;
}

extern __shared__ unsigned int lds_u[];

__global__ void __launch_bounds__(1024) vq_main(
    const float* __restrict__ x,      // [65536][64]
    const float* __restrict__ emb,    // [2048][64]
    float* __restrict__ q_out,        // [65536][64]
    float* __restrict__ enc_out,      // [65536][2048]
    float* __restrict__ counts,       // ws [2048]
    float* __restrict__ dw,           // ws [2048][64]
    float* __restrict__ loss_acc)     // ws [1]
{
    float* lds_f = (float*)lds_u;
    const int tid = threadIdx.x;

    // ---- Stage embedding into LDS as bf16, compute e_sq in f32 ----
    for (int k = tid; k < K_EMB; k += 1024) {
        const float* er = emb + k * D_DIM;
        float ss = 0.0f;
        #pragma unroll
        for (int j = 0; j < 32; ++j) {
            float a = er[2 * j], b = er[2 * j + 1];
            ss += a * a + b * b;
            lds_u[k * ROW_U + j] = pack_bf16x2(a, b);
        }
        lds_f[ESQ_OFF + k] = ss;
    }
    __syncthreads();

    const int wave = tid >> 5;
    const int lane = tid & 31;
    const int nlo  = lane & 15;
    const int h    = lane >> 4;
    const int rowBase = blockIdx.x * 512 + wave * 16;

    // ---- A fragments: 16 rows x 64 K (bf16), resident in registers ----
    // A 16x32 bf16 in this code's contiguous k order (same as B, not AMD's table order): lane holds row M = nlo;
//   VGPR j (0..7) -> K = 16h + 2j, 16h + 2j + 1
    Frag afrag[2];
    const float* xr = x + (size_t)(rowBase + nlo) * D_DIM;
    #pragma unroll
    for (int c = 0; c < 4; ++c) {
        int s  = 32 * (c >> 1) + 16 * h + 8 * (c & 1);
        int f  = c >> 1;
        int jb = (c & 1) * 4;
        #pragma unroll
        for (int j2 = 0; j2 < 4; ++j2)
            afrag[f].u[jb + j2] = pack_bf16x2(xr[s + 2 * j2], xr[s + 2 * j2 + 1]);
    }

    float bestv[8];
    int   bestk[8];
    #pragma unroll
    for (int r = 0; r < 8; ++r) { bestv[r] = 3.4e38f; bestk[r] = 0; }

    // ---- Sweep all 128 column tiles: dist(k) = e_sq[k] - 2 * <x, e_k> ----
    // Unroll x2 so tile i's argmin VALU fills tile i+1's WMMA hazard shadow.
    #pragma unroll 2
    for (int k0 = 0; k0 < K_EMB; k0 += 16) {
        // B 32x16 bf16 layout: lane holds column N = nlo, K split by half:
        //   frag0 VGPR j -> d = 16h + 2j   (LDS dword 8h + j)
        //   frag1 VGPR j -> d = 32 + 16h + 2j
        Frag b0, b1;
        const unsigned int* brow = lds_u + (size_t)(k0 + nlo) * ROW_U;
        #pragma unroll
        for (int j = 0; j < 8; ++j) {
            b0.u[j] = brow[8 * h + j];
            b1.u[j] = brow[16 + 8 * h + j];
        }
        v8f c = {0.f, 0.f, 0.f, 0.f, 0.f, 0.f, 0.f, 0.f};
        c = __builtin_amdgcn_wmma_f32_16x16x32_bf16(false, afrag[0].v, false, b0.v,
                                                    (short)0, c, false, false);
        c = __builtin_amdgcn_wmma_f32_16x16x32_bf16(false, afrag[1].v, false, b1.v,
                                                    (short)0, c, false, false);
        int   k   = k0 + nlo;
        float esq = lds_f[ESQ_OFF + k];
        #pragma unroll
        for (int r = 0; r < 8; ++r) {   // C layout: VGPR r -> row 8h + r, col nlo
            float val = esq - 2.0f * c[r];
            if (val < bestv[r]) { bestv[r] = val; bestk[r] = k; }
        }
    }

    // ---- Argmin across the 16 lanes of each half (columns), ties -> low k ----
    #pragma unroll
    for (int r = 0; r < 8; ++r) {
        float v  = bestv[r];
        int   ki = bestk[r];
        #pragma unroll
        for (int off = 1; off < 16; off <<= 1) {
            float ov = __shfl_xor(v, off, 32);
            int   oi = __shfl_xor(ki, off, 32);
            if (ov < v || (ov == v && oi < ki)) { v = ov; ki = oi; }
        }
        bestv[r] = v; bestk[r] = ki;
    }

    // ---- Writeback: quantized, one-hot, dw atomics, counts, loss ----
    float lacc = 0.0f;
    const v4f zero4 = {0.0f, 0.0f, 0.0f, 0.0f};
    #pragma unroll
    for (int r = 0; r < 8; ++r) {
        int row = rowBase + 8 * h + r;
        int idx = bestk[r];

        const float4 e4 = *(const float4*)(emb + (size_t)idx * D_DIM + nlo * 4);
        const float4 x4 = *(const float4*)(x + (size_t)row * D_DIM + nlo * 4);
        v4f q4 = {e4.x, e4.y, e4.z, e4.w};
        __builtin_nontemporal_store(q4, (v4f*)(q_out + (size_t)row * D_DIM + nlo * 4));

        float d0 = e4.x - x4.x, d1 = e4.y - x4.y, d2 = e4.z - x4.z, d3 = e4.w - x4.w;
        lacc += d0 * d0 + d1 * d1 + d2 * d2 + d3 * d3;

        float* dwr = dw + (size_t)idx * D_DIM + nlo * 4;
        atomicAdd(dwr + 0, x4.x);
        atomicAdd(dwr + 1, x4.y);
        atomicAdd(dwr + 2, x4.z);
        atomicAdd(dwr + 3, x4.w);
        if (nlo == 0) atomicAdd(counts + idx, 1.0f);

        // One-hot row: stream zeros (non-temporal), then one lane stores 1.0.
        // Same-wave stores to the same address are kept in order (ISA ch.5).
        float* er = enc_out + (size_t)row * K_EMB;
        #pragma unroll 4
        for (int j = 0; j < 32; ++j) {          // half-wave covers 2048 floats
            int kb = (j * 16 + nlo) * 4;        // lanes contiguous per store
            __builtin_nontemporal_store(zero4, (v4f*)(er + kb));
        }
        if (nlo == 0) er[idx] = 1.0f;
    }
    #pragma unroll
    for (int off = 1; off < 32; off <<= 1) lacc += __shfl_xor(lacc, off, 32);
    if (lane == 0) atomicAdd(loss_acc, lacc);
}

__global__ void __launch_bounds__(1024) vq_final(
    const float* __restrict__ emb,
    const float* __restrict__ cs_in,
    const float* __restrict__ counts,
    const float* __restrict__ dw,
    const float* __restrict__ loss_acc,
    float* __restrict__ out_loss,
    float* __restrict__ out_perp,
    float* __restrict__ out_emb,
    float* __restrict__ out_cs)
{
    __shared__ float red[64];
    __shared__ float nshared;
    const int tid = threadIdx.x;

    float nsum = 0.0f, plog = 0.0f;
    float ncs_local[2];
    #pragma unroll
    for (int i = 0; i < 2; ++i) {
        int   k   = tid + i * 1024;
        float cnt = counts[k];
        float ncs = DECAYF * cs_in[k] + OMDF * cnt;
        out_cs[k]    = ncs;
        ncs_local[i] = ncs;
        nsum += ncs;
        float p = cnt / (float)N_ROWS;
        plog += p * logf(p + 1e-10f);
    }
    #pragma unroll
    for (int off = 1; off < 32; off <<= 1) {
        nsum += __shfl_xor(nsum, off, 32);
        plog += __shfl_xor(plog, off, 32);
    }
    int lane = tid & 31, wave = tid >> 5;
    if (lane == 0) { red[wave] = nsum; red[32 + wave] = plog; }
    __syncthreads();
    if (tid == 0) {
        float n = 0.0f, pl = 0.0f;
        for (int w = 0; w < 32; ++w) { n += red[w]; pl += red[32 + w]; }
        nshared     = n;
        out_perp[0] = expf(-pl);
        out_loss[0] = loss_acc[0] / ((float)N_ROWS * (float)D_DIM);
    }
    __syncthreads();
    float n = nshared;
    #pragma unroll
    for (int i = 0; i < 2; ++i) {
        int   k   = tid + i * 1024;
        float cs  = (ncs_local[i] + EPSV) / (n + (float)K_EMB * EPSV) * n;
        float inv = OMDF / cs;
        for (int d = 0; d < D_DIM; d += 4) {
            float4 e4 = *(const float4*)(emb + (size_t)k * D_DIM + d);
            float4 w4 = *(const float4*)(dw  + (size_t)k * D_DIM + d);
            float4 o;
            o.x = DECAYF * e4.x + w4.x * inv;
            o.y = DECAYF * e4.y + w4.y * inv;
            o.z = DECAYF * e4.z + w4.z * inv;
            o.w = DECAYF * e4.w + w4.w * inv;
            *(float4*)(out_emb + (size_t)k * D_DIM + d) = o;
        }
    }
}

extern "C" void kernel_launch(void* const* d_in, const int* in_sizes, int n_in,
                              void* d_out, int out_size, void* d_ws, size_t ws_size,
                              hipStream_t stream) {
    const float* x     = (const float*)d_in[0];  // (65536, 64)
    const float* emb   = (const float*)d_in[1];  // (1, 2048, 64)
    const float* cs_in = (const float*)d_in[2];  // (1, 2048)

    float* out      = (float*)d_out;
    float* q_out    = out;                                   // 65536*64
    float* enc_out  = out + (size_t)N_ROWS * D_DIM;          // 65536*2048
    float* out_loss = enc_out + (size_t)N_ROWS * K_EMB;      // 1
    float* out_perp = out_loss + 1;                          // 1
    float* out_emb  = out_perp + 1;                          // 2048*64
    float* out_cs   = out_emb + (size_t)K_EMB * D_DIM;       // 2048

    float* ws       = (float*)d_ws;
    float* counts   = ws;                                    // 2048
    float* dwacc    = ws + K_EMB;                            // 2048*64
    float* loss_acc = dwacc + (size_t)K_EMB * D_DIM;         // 1

    int zn = K_EMB + K_EMB * D_DIM + 1;
    hipLaunchKernelGGL(vq_init, dim3((zn + 255) / 256), dim3(256), 0, stream, ws, zn);

    size_t lds_bytes = (size_t)K_EMB * ROW_U * 4 + (size_t)K_EMB * 4;  // 303,104 B
    hipLaunchKernelGGL(vq_main, dim3(128), dim3(1024), lds_bytes, stream,
                       x, emb, q_out, enc_out, counts, dwacc, loss_acc);

    hipLaunchKernelGGL(vq_final, dim3(1), dim3(1024), 0, stream,
                       emb, cs_in, counts, dwacc, loss_acc,
                       out_loss, out_perp, out_emb, out_cs);
}
